// EnhancedGraphConv_6030134083772
// MI455X (gfx1250) — matrix operand order repaired and checked by reading, not compiled, not run on hardware
//
#include <hip/hip_runtime.h>

typedef __attribute__((ext_vector_type(16))) _Float16 v16h;
typedef __attribute__((ext_vector_type(8)))  float    v8f;

namespace {

constexpr int Bb = 2, Nn = 640, Cc = 64, Oo = 64, Ee = 18;
constexpr int NT = Nn / 16;     // 40 j-tiles of 16 rows
constexpr int WV = 4;           // waves per workgroup (wave32)

// ---- workspace layout ----
// packed f16 B-layout weight tiles (half-element offsets)
constexpr size_t H_pWe1 = 0;                       //  32x64
constexpr size_t H_pWe2 = H_pWe1 + 32 * 64;        //  64x64
constexpr size_t H_pWe3 = H_pWe2 + 64 * 64;        //  64x32
constexpr size_t H_pWa1 = H_pWe3 + 64 * 32;        // 160x64
constexpr size_t H_pWa2 = H_pWa1 + 160 * 64;       //  64x32
constexpr size_t H_pWg1 = H_pWa2 + 64 * 32;        //  96x64
constexpr size_t H_pWg2 = H_pWg1 + 96 * 64;        //  64x64
constexpr size_t H_xh   = H_pWg2 + 64 * 64;        // f16 copy of x [B*N,64]
constexpr size_t H_end  = H_xh + (size_t)Bb * Nn * Cc;
// f32 region (float-element offsets); H_end*2 bytes is 16B aligned
constexpr size_t F_tn = (H_end * 2) / 4;                       // tn  [B*N,64]
constexpr size_t F_sf = F_tn + (size_t)Bb * Nn * Oo;           // sf  [B*N,64]

// ---------------- WMMA helpers ----------------

__device__ inline v8f wmma_f16(v16h a, v16h b, v8f c) {
  return __builtin_amdgcn_wmma_f32_16x16x32_f16(false, a, false, b, (short)0, c,
                                                false, false);
}

// Gather a 16x32 f16 A tile from an LDS row-major [16][ldK] buffer.
// 16-bit A layout: lanes 0-15 row M=lane, VGPR v holds K pair
//   kk = 2v + (v>=4 ? 8 : 0) + 8*(lane>=16)  ... per ISA 7.12.2 table.
// The two 4-VGPR K-runs are byte-contiguous -> compiler emits 2x ds_load_b128.
__device__ inline v16h gatherA_s(const _Float16* lds, int ldK, int kbase, int lane) {
  const int M = lane & 15, hh = lane >> 4;
  union { v16h h; unsigned u[8]; } r;
#pragma unroll
  for (int v = 0; v < 8; ++v) {
    const int kk = 2 * v + ((v >= 4) ? 8 : 0) + 8 * hh;
    r.u[v] = *(const unsigned*)(lds + M * ldK + kbase + kk);
  }
  return r.h;
}

// Same-gather but every row M is the same vector (broadcast xi row), straight
// from a global f16 row (uniform address per lane-half -> broadcast friendly).
__device__ inline v16h gatherA_bcast(const _Float16* row, int kbase, int lane) {
  const int hh = lane >> 4;
  union { v16h h; unsigned u[8]; } r;
#pragma unroll
  for (int v = 0; v < 8; ++v) {
    const int kk = 2 * v + ((v >= 4) ? 8 : 0) + 8 * hh;
    r.u[v] = *(const unsigned*)(row + kbase + kk);
  }
  return r.h;
}

// Load a pre-packed 32x16 B tile: 512 halfs per tile, lane-contiguous 32B
// (two clause-able global_load_b128 per lane; resident in WGP$/L2).
__device__ inline v16h loadB(const uint4* packed, int tileIdx, int lane) {
  union { v16h h; uint4 q[2]; } r;
  r.q[0] = packed[(size_t)tileIdx * 64 + lane * 2 + 0];
  r.q[1] = packed[(size_t)tileIdx * 64 + lane * 2 + 1];
  return r.h;
}

// Store a 16x16 f32 C tile to LDS f16 with bias+ReLU.
// C layout: lane holds column n=lane&15; VGPR v holds row v + 8*(lane>=16).
__device__ inline void store_act_relu(_Float16* buf, int ldN, int nt, int lane,
                                      v8f c, float bias) {
  const int nn = lane & 15, hh = lane >> 4;
#pragma unroll
  for (int v = 0; v < 8; ++v)
    buf[(v + 8 * hh) * ldN + nt * 16 + nn] = (_Float16)fmaxf(c[v] + bias, 0.f);
}

// ---------------- pack kernel ----------------
// Pack W[K x Ncols] f32 into B-layout f16 tiles (k zero-padded to Kpad).
// Tile (kc,nt): lane l holds halfs q=0..15 with k = kc*32 + (l>=16?8:0) + (q<8?q:8+q),
// n = nt*16 + (l&15); stored lane-contiguous.
__global__ void pack_b(const float* __restrict__ src, _Float16* __restrict__ dst,
                       int K, int Kpad, int Ncols) {
  const int idx = blockIdx.x * blockDim.x + threadIdx.x;
  const int total = Kpad * Ncols;
  if (idx >= total) return;
  const int tile = idx >> 9;
  const int rem = idx & 511;
  const int lane = rem >> 4;
  const int q = rem & 15;
  const int ntn = Ncols >> 4;
  const int kc = tile / ntn, nt = tile - kc * ntn;
  const int k = kc * 32 + ((lane >= 16) ? 8 : 0) + ((q < 8) ? q : 8 + q);
  const int n = nt * 16 + (lane & 15);
  const float v = (k < K) ? src[(size_t)k * Ncols + n] : 0.f;
  dst[idx] = (_Float16)v;
}

// ---------------- node precompute ----------------
// sf = x@Ws+bs, tn = x@Wn+bn, xh = f16(x). Tiny (1280x(64x64x2) MACs).
__global__ __launch_bounds__(64) void node_kernel(
    const float* __restrict__ x, const float* __restrict__ Ws,
    const float* __restrict__ bs, const float* __restrict__ Wn,
    const float* __restrict__ bn, float* __restrict__ sf,
    float* __restrict__ tn, _Float16* __restrict__ xh) {
  const int row = blockIdx.x;
  const int t = threadIdx.x;
  __shared__ float xr[64];
  xr[t] = x[(size_t)row * 64 + t];
  __syncthreads();
  xh[(size_t)row * 64 + t] = (_Float16)xr[t];
  float a = bs[t], c = bn[t];
#pragma unroll 8
  for (int k = 0; k < 64; ++k) {
    const float xv = xr[k];
    a += xv * Ws[k * 64 + t];
    c += xv * Wn[k * 64 + t];
  }
  sf[(size_t)row * 64 + t] = a;
  tn[(size_t)row * 64 + t] = c;
}

// ---------------- fused main kernel ----------------

struct GParams {
  const float *adj, *ef;
  const float *be1, *be2, *be3, *ba1, *ba2, *Wa3, *ba3, *bg1, *bg2;
  const float *Wc1, *bc1, *Wc2, *bc2;
  const _Float16 *pWe1, *pWe2, *pWe3, *pWa1, *pWa2, *pWg1, *pWg2, *xh;
  const float *tn, *sf;
  float* out;
};

__global__ __launch_bounds__(WV * 32) void graph_main(GParams p) {
  const int bi = blockIdx.x;                  // (b, i)
  const int b = bi / Nn, i = bi - b * Nn;
  const int lane = threadIdx.x & 31;
  const int w = threadIdx.x >> 5;
  const int nn = lane & 15, hh = lane >> 4;

  __shared__ alignas(16) _Float16 s_ef[WV][16 * 32];
  __shared__ alignas(16) _Float16 s_xj[WV][16 * 64];
  __shared__ alignas(16) _Float16 s_pe[WV][16 * 32];
  __shared__ alignas(16) _Float16 s_b1[WV][16 * 64];
  __shared__ alignas(16) _Float16 s_b2[WV][16 * 64];
  __shared__ float s_msg[WV][64];
  __shared__ float s_m[WV], s_s[WV];
  __shared__ float s_comb[2 * Oo];
  __shared__ float s_h1[Oo];

  const uint4* qWe1 = (const uint4*)p.pWe1;
  const uint4* qWe2 = (const uint4*)p.pWe2;
  const uint4* qWe3 = (const uint4*)p.pWe3;
  const uint4* qWa1 = (const uint4*)p.pWa1;
  const uint4* qWa2 = (const uint4*)p.pWa2;
  const uint4* qWg1 = (const uint4*)p.pWg1;
  const uint4* qWg2 = (const uint4*)p.pWg2;

  // hoisted per-lane bias values (column = t*16 + nn)
  float be1v[4], be2v[4], ba1v[4], bg1v[4], bg2v[4];
#pragma unroll
  for (int t = 0; t < 4; ++t) {
    be1v[t] = p.be1[t * 16 + nn];
    be2v[t] = p.be2[t * 16 + nn];
    ba1v[t] = p.ba1[t * 16 + nn];
    bg1v[t] = p.bg1[t * 16 + nn];
    bg2v[t] = p.bg2[t * 16 + nn];
  }
  float be3v[2], ba2v[2];
#pragma unroll
  for (int t = 0; t < 2; ++t) {
    be3v[t] = p.be3[t * 16 + nn];
    ba2v[t] = p.ba2[t * 16 + nn];
  }
  const float wa3a = p.Wa3[nn], wa3b = p.Wa3[16 + nn], ba3s = p.ba3[0];

  // xi broadcast A tiles (loop invariant)
  const _Float16* xir = p.xh + (size_t)(b * Nn + i) * Cc;
  const v16h xiA0 = gatherA_bcast(xir, 0, lane);
  const v16h xiA1 = gatherA_bcast(xir, 32, lane);

  const float* adjr = p.adj + ((size_t)b * Nn + i) * Nn;
  const float* efbase = p.ef + (size_t)(b * Nn + i) * Nn * Ee;
  const float* tnbase = p.tn + (size_t)b * Nn * Oo;

  _Float16* efL = s_ef[w];
  _Float16* xjL = s_xj[w];
  _Float16* peL = s_pe[w];
  _Float16* b1L = s_b1[w];
  _Float16* b2L = s_b2[w];

  // prefetch this wave's first edge/tn tiles into cache before compute starts
  {
    const int jb0 = w * 16;
    __builtin_prefetch(efbase + (size_t)jb0 * Ee + lane * 9, 0, 0);
    __builtin_prefetch(tnbase + (size_t)jb0 * Oo + lane * 32, 0, 0);
  }

  // online (streaming) masked softmax state
  float m_run = -1e30f, s_run = 0.f;
  float acc[4] = {0.f, 0.f, 0.f, 0.f};
  const v8f zv = {};

  for (int jt = w; jt < NT; jt += WV) {
    const int jb = jt * 16;

    // ---- prefetch NEXT iteration's HBM-streamed tiles (pure hint, no cnt) --
    if (jt + WV < NT) {
      const size_t jbn = (size_t)(jb + WV * 16);
      // edge tile: 16 rows x 18 f32 = 1152B contiguous; 32 lanes x 36B covers it
      __builtin_prefetch(efbase + jbn * Ee + lane * 9, 0, 0);
      // tn tile: 16 rows x 64 f32 = 4096B contiguous; 32 lanes x 128B
      __builtin_prefetch(tnbase + jbn * Oo + lane * 32, 0, 0);
    }

    // ---- stage edge-feature tile [16][32] (K zero-padded 18->32) ----
    {
      const float* efr = efbase + (size_t)jb * Ee;
      for (int idx = lane; idx < 16 * 32; idx += 32) {
        const int r = idx >> 5, k = idx & 31;
        const float vv = (k < Ee) ? efr[r * Ee + k] : 0.f;
        efL[idx] = (_Float16)vv;
      }
      // ---- stage xj tile [16][64] from f16 x copy ----
      const uint4* src = (const uint4*)(p.xh + (size_t)(b * Nn + jb) * Cc);
      uint4* dst = (uint4*)xjL;
      for (int idx = lane; idx < 128; idx += 32) dst[idx] = src[idx];
    }
    __builtin_amdgcn_wave_barrier();

    // ======== edge MLP: 18(->32) -> 64 -> 64 -> 32 ========
    v8f c1[4];
    {
      const v16h a = gatherA_s(efL, 32, 0, lane);
#pragma unroll
      for (int t = 0; t < 4; ++t) c1[t] = wmma_f16(a, loadB(qWe1, t, lane), zv);
    }
#pragma unroll
    for (int t = 0; t < 4; ++t) store_act_relu(b1L, 64, t, lane, c1[t], be1v[t]);
    __builtin_amdgcn_wave_barrier();

    v8f c2[4];
#pragma unroll
    for (int t = 0; t < 4; ++t) c2[t] = zv;
#pragma unroll
    for (int kc = 0; kc < 2; ++kc) {
      const v16h a = gatherA_s(b1L, 64, 32 * kc, lane);
#pragma unroll
      for (int t = 0; t < 4; ++t)
        c2[t] = wmma_f16(a, loadB(qWe2, kc * 4 + t, lane), c2[t]);
    }
#pragma unroll
    for (int t = 0; t < 4; ++t) store_act_relu(b2L, 64, t, lane, c2[t], be2v[t]);
    __builtin_amdgcn_wave_barrier();

    v8f c3[2];
    c3[0] = zv; c3[1] = zv;
#pragma unroll
    for (int kc = 0; kc < 2; ++kc) {
      const v16h a = gatherA_s(b2L, 64, 32 * kc, lane);
#pragma unroll
      for (int t = 0; t < 2; ++t)
        c3[t] = wmma_f16(a, loadB(qWe3, kc * 2 + t, lane), c3[t]);
    }
#pragma unroll
    for (int t = 0; t < 2; ++t) store_act_relu(peL, 32, t, lane, c3[t], be3v[t]);
    __builtin_amdgcn_wave_barrier();

    const v16h axj0 = gatherA_s(xjL, 64, 0, lane);
    const v16h axj1 = gatherA_s(xjL, 64, 32, lane);
    const v16h ape  = gatherA_s(peL, 32, 0, lane);

    // ======== attention MLP: [xi|xj|pe](160) -> 64 -> 32 -> score ========
    v8f h1[4];
#pragma unroll
    for (int t = 0; t < 4; ++t) h1[t] = zv;
#pragma unroll
    for (int t = 0; t < 4; ++t) h1[t] = wmma_f16(xiA0, loadB(qWa1, 0 + t, lane), h1[t]);
#pragma unroll
    for (int t = 0; t < 4; ++t) h1[t] = wmma_f16(xiA1, loadB(qWa1, 4 + t, lane), h1[t]);
#pragma unroll
    for (int t = 0; t < 4; ++t) h1[t] = wmma_f16(axj0, loadB(qWa1, 8 + t, lane), h1[t]);
#pragma unroll
    for (int t = 0; t < 4; ++t) h1[t] = wmma_f16(axj1, loadB(qWa1, 12 + t, lane), h1[t]);
#pragma unroll
    for (int t = 0; t < 4; ++t) h1[t] = wmma_f16(ape, loadB(qWa1, 16 + t, lane), h1[t]);
#pragma unroll
    for (int t = 0; t < 4; ++t) store_act_relu(b1L, 64, t, lane, h1[t], ba1v[t]);
    __builtin_amdgcn_wave_barrier();

    v8f h2[2];
    h2[0] = zv; h2[1] = zv;
#pragma unroll
    for (int kc = 0; kc < 2; ++kc) {
      const v16h a = gatherA_s(b1L, 64, 32 * kc, lane);
#pragma unroll
      for (int t = 0; t < 2; ++t)
        h2[t] = wmma_f16(a, loadB(qWa2, kc * 2 + t, lane), h2[t]);
    }
#pragma unroll
    for (int t = 0; t < 2; ++t)
#pragma unroll
      for (int v = 0; v < 8; ++v) h2[t][v] = fmaxf(h2[t][v] + ba2v[t], 0.f);

    // per-row scores: reduce h2 rows (C layout) across lane columns
    float sc[8], wmask[8];
#pragma unroll
    for (int v = 0; v < 8; ++v) {
      float pp = h2[0][v] * wa3a + h2[1][v] * wa3b;
      pp += __shfl_xor(pp, 1, 32);
      pp += __shfl_xor(pp, 2, 32);
      pp += __shfl_xor(pp, 4, 32);
      pp += __shfl_xor(pp, 8, 32);
      sc[v] = pp + ba3s;
    }
#pragma unroll
    for (int v = 0; v < 8; ++v) {
      const float av = adjr[jb + v + 8 * hh];
      wmask[v] = (av > 0.f) ? 1.f : 0.f;
      sc[v] = (av > 0.f) ? sc[v] : -1e30f;
    }

    // ======== gate MLP: [xj|pe](96) -> 64 -> 64 -> sigmoid ========
    v8f g1[4];
#pragma unroll
    for (int t = 0; t < 4; ++t) g1[t] = zv;
#pragma unroll
    for (int t = 0; t < 4; ++t) g1[t] = wmma_f16(axj0, loadB(qWg1, 0 + t, lane), g1[t]);
#pragma unroll
    for (int t = 0; t < 4; ++t) g1[t] = wmma_f16(axj1, loadB(qWg1, 4 + t, lane), g1[t]);
#pragma unroll
    for (int t = 0; t < 4; ++t) g1[t] = wmma_f16(ape, loadB(qWg1, 8 + t, lane), g1[t]);
#pragma unroll
    for (int t = 0; t < 4; ++t) store_act_relu(b2L, 64, t, lane, g1[t], bg1v[t]);
    __builtin_amdgcn_wave_barrier();

    v8f g2[4];
#pragma unroll
    for (int t = 0; t < 4; ++t) g2[t] = zv;
#pragma unroll
    for (int kc = 0; kc < 2; ++kc) {
      const v16h a = gatherA_s(b2L, 64, 32 * kc, lane);
#pragma unroll
      for (int t = 0; t < 4; ++t)
        g2[t] = wmma_f16(a, loadB(qWg2, kc * 4 + t, lane), g2[t]);
    }
#pragma unroll
    for (int t = 0; t < 4; ++t)
#pragma unroll
      for (int v = 0; v < 8; ++v)
        g2[t][v] = 1.f / (1.f + __expf(-(g2[t][v] + bg2v[t])));

    // ======== streaming masked softmax + message accumulation ========
    float tmax = -1e30f;
#pragma unroll
    for (int v = 0; v < 8; ++v) tmax = fmaxf(tmax, sc[v]);
    tmax = fmaxf(tmax, __shfl_xor(tmax, 16, 32));
    const float m_new = fmaxf(m_run, tmax);
    const float scale = __expf(m_run - m_new);
    s_run *= scale;
#pragma unroll
    for (int t = 0; t < 4; ++t) acc[t] *= scale;

    float wv[8];
    float hsum = 0.f;
#pragma unroll
    for (int v = 0; v < 8; ++v) {
      wv[v] = wmask[v] * __expf(sc[v] - m_new);
      hsum += wv[v];
    }
    s_run += hsum + __shfl_xor(hsum, 16, 32);
    m_run = m_new;

#pragma unroll
    for (int t = 0; t < 4; ++t) {
      float a2 = acc[t];
#pragma unroll
      for (int v = 0; v < 8; ++v) {
        const float tnv = tnbase[(size_t)(jb + v + 8 * hh) * Oo + t * 16 + nn];
        a2 += wv[v] * g2[t][v] * tnv;
      }
      acc[t] = a2;
    }
  }

  // ---- merge the 4 per-wave softmax states ----
#pragma unroll
  for (int t = 0; t < 4; ++t) {
    const float a2 = acc[t] + __shfl_xor(acc[t], 16, 32);
    if (lane < 16) s_msg[w][t * 16 + lane] = a2;
  }
  if (lane == 0) { s_m[w] = m_run; s_s[w] = s_run; }
  __syncthreads();

  const int t = threadIdx.x;
  if (t < Oo) {
    float M = s_m[0];
    for (int q = 1; q < WV; ++q) M = fmaxf(M, s_m[q]);
    float S = 0.f, msg = 0.f;
    for (int q = 0; q < WV; ++q) {
      const float e = __expf(s_m[q] - M);
      S += s_s[q] * e;
      msg += s_msg[q][t] * e;
    }
    msg = (S > 0.f) ? msg / S : 0.f;
    s_comb[t] = p.sf[(size_t)(b * Nn + i) * Oo + t];
    s_comb[Oo + t] = msg;
  }
  __syncthreads();
  if (t < Oo) {
    float h = p.bc1[t];
    for (int k = 0; k < 2 * Oo; ++k) h += s_comb[k] * p.Wc1[k * Oo + t];
    s_h1[t] = fmaxf(h, 0.f);
  }
  __syncthreads();
  if (t < Oo) {
    float o = p.bc2[t];
    for (int k = 0; k < Oo; ++k) o += s_h1[k] * p.Wc2[k * Oo + t];
    p.out[(size_t)(b * Nn + i) * Oo + t] = o;
  }
}

}  // namespace

extern "C" void kernel_launch(void* const* d_in, const int* in_sizes, int n_in,
                              void* d_out, int out_size, void* d_ws, size_t ws_size,
                              hipStream_t stream) {
  (void)in_sizes; (void)n_in; (void)out_size; (void)ws_size;
  const float* x   = (const float*)d_in[0];
  const float* adj = (const float*)d_in[1];
  const float* ef  = (const float*)d_in[2];
  const float* Ws  = (const float*)d_in[3];
  const float* bs  = (const float*)d_in[4];
  const float* Wn  = (const float*)d_in[5];
  const float* bn  = (const float*)d_in[6];
  const float* We1 = (const float*)d_in[7];
  const float* be1 = (const float*)d_in[8];
  const float* We2 = (const float*)d_in[9];
  const float* be2 = (const float*)d_in[10];
  const float* We3 = (const float*)d_in[11];
  const float* be3 = (const float*)d_in[12];
  const float* Wa1 = (const float*)d_in[13];
  const float* ba1 = (const float*)d_in[14];
  const float* Wa2 = (const float*)d_in[15];
  const float* ba2 = (const float*)d_in[16];
  const float* Wa3 = (const float*)d_in[17];
  const float* ba3 = (const float*)d_in[18];
  const float* Wg1 = (const float*)d_in[19];
  const float* bg1 = (const float*)d_in[20];
  const float* Wg2 = (const float*)d_in[21];
  const float* bg2 = (const float*)d_in[22];
  const float* Wc1 = (const float*)d_in[23];
  const float* bc1 = (const float*)d_in[24];
  const float* Wc2 = (const float*)d_in[25];
  const float* bc2 = (const float*)d_in[26];

  _Float16* wsH = (_Float16*)d_ws;
  float* wsF = (float*)d_ws;

  // 1) pack weights into B-layout f16 tiles
  auto pack = [&](const float* src, size_t hOff, int K, int Kpad, int Ncols) {
    const int total = Kpad * Ncols;
    pack_b<<<(total + 255) / 256, 256, 0, stream>>>(src, wsH + hOff, K, Kpad, Ncols);
  };
  pack(We1, H_pWe1, 18, 32, 64);
  pack(We2, H_pWe2, 64, 64, 64);
  pack(We3, H_pWe3, 64, 64, 32);
  pack(Wa1, H_pWa1, 160, 160, 64);
  pack(Wa2, H_pWa2, 64, 64, 32);
  pack(Wg1, H_pWg1, 96, 96, 64);
  pack(Wg2, H_pWg2, 64, 64, 64);

  // 2) node transforms + f16 copy of x
  node_kernel<<<Bb * Nn, 64, 0, stream>>>(x, Ws, bs, Wn, bn,
                                          wsF + F_sf, wsF + F_tn, wsH + H_xh);

  // 3) fused per-node kernel
  GParams p;
  p.adj = adj; p.ef = ef;
  p.be1 = be1; p.be2 = be2; p.be3 = be3;
  p.ba1 = ba1; p.ba2 = ba2; p.Wa3 = Wa3; p.ba3 = ba3;
  p.bg1 = bg1; p.bg2 = bg2;
  p.Wc1 = Wc1; p.bc1 = bc1; p.Wc2 = Wc2; p.bc2 = bc2;
  p.pWe1 = wsH + H_pWe1; p.pWe2 = wsH + H_pWe2; p.pWe3 = wsH + H_pWe3;
  p.pWa1 = wsH + H_pWa1; p.pWa2 = wsH + H_pWa2;
  p.pWg1 = wsH + H_pWg1; p.pWg2 = wsH + H_pWg2;
  p.xh = wsH + H_xh;
  p.tn = wsF + F_tn; p.sf = wsF + F_sf;
  p.out = (float*)d_out;

  graph_main<<<Bb * Nn, WV * 32, 0, stream>>>(p);
}
